// Seq2Seq_65644280152152
// MI455X (gfx1250) — compile-verified, matrix operand order checked by reading, not run on hardware
//
#include <hip/hip_runtime.h>
#include <hip/hip_bf16.h>
#include <math.h>

// Problem constants
constexpr int CB = 16, CS = 400, CT = 64, CE = 128, CH = 256, CV = 50000;
constexpr int CVE = CV + 50;     // extended vocab
constexpr int G3  = 3 * CH;      // 768 gate width

typedef __attribute__((ext_vector_type(16))) _Float16 v16h;
typedef __attribute__((ext_vector_type(8)))  float    v8f;

// ---------------------------------------------------------------------------
// WMMA helpers (CDNA5 wave32, V_WMMA_F32_16X16X32_F16)
// ---------------------------------------------------------------------------
__device__ __forceinline__ v16h frag_load(const _Float16* p, int ld, int lane) {
  const _Float16* row = p + (size_t)(lane & 15) * ld + ((lane & 16) ? 8 : 0);
  v16h f;
#pragma unroll
  for (int j = 0; j < 4; ++j) {
    f[2 * j]       = row[2 * j];
    f[2 * j + 1]   = row[2 * j + 1];
    f[8 + 2 * j]   = row[16 + 2 * j];
    f[8 + 2 * j + 1] = row[16 + 2 * j + 1];
  }
  return f;
}

__device__ __forceinline__ v8f wmma_acc(v16h a, v16h b, v8f c) {
  return __builtin_amdgcn_wmma_f32_16x16x32_f16(false, a, false, b, (short)0, c,
                                                false, false);
}

__device__ __forceinline__ float sigmf(float x) { return 1.f / (1.f + __expf(-x)); }

// ---------------------------------------------------------------------------
// Generic WMMA GEMM: C[MxN] = A[MxK] @ W[NxK]^T + bias, optional f32 and f16
// outputs, optional leaky-relu. One wave per 16x16 tile, 4 waves per block.
// ---------------------------------------------------------------------------
__global__ void k_gemm(const _Float16* __restrict__ A, int lda,
                       const _Float16* __restrict__ Bw, int ldb,
                       const float* __restrict__ bias,
                       float* __restrict__ C, int ldc,
                       _Float16* __restrict__ D16, int ld16,
                       int M, int N, int K, int act) {
  int wave = threadIdx.x >> 5;
  int lane = threadIdx.x & 31;
  int n0 = (blockIdx.x * (blockDim.x >> 5) + wave) * 16;
  int m0 = blockIdx.y * 16;
  if (n0 >= N || m0 >= M) return;

  float bv = bias ? bias[n0 + (lane & 15)] : 0.f;
  v8f acc;
#pragma unroll
  for (int j = 0; j < 8; ++j) acc[j] = bv;

  for (int k = 0; k < K; k += 32) {
    v16h a = frag_load(A + (size_t)m0 * lda + k, lda, lane);
    v16h b = frag_load(Bw + (size_t)n0 * ldb + k, ldb, lane);
    acc = wmma_acc(a, b, acc);
  }

  int n = lane & 15;
  int mh = (lane & 16) ? 8 : 0;
#pragma unroll
  for (int j = 0; j < 8; ++j) {
    int m = m0 + mh + j;
    float v = acc[j];
    if (act == 1) v = (v > 0.f) ? v : 0.01f * v;   // leaky_relu 0.01
    if (C)   C[(size_t)m * ldc + n0 + n] = v;
    if (D16) D16[(size_t)m * ld16 + n0 + n] = (_Float16)v;
  }
}

// ---------------------------------------------------------------------------
// Vocab GEMM: logits[16,V] = vtmp[16,256] @ vhat[V,256]^T + b.
// The 8KB A matrix is staged block-wide into LDS with CDNA5 async-DMA
// (GLOBAL_LOAD_ASYNC_TO_LDS_B128 + s_wait_asynccnt), then every one of the
// 8 waves/block reads its A fragments from LDS while streaming B from the
// L2-resident f16 weight copy directly into WMMA.
// ---------------------------------------------------------------------------
__global__ void k_gemm_vocab(const _Float16* __restrict__ A,   // [16,256]
                             const _Float16* __restrict__ Bw,  // [V,256]
                             const float* __restrict__ bias,
                             float* __restrict__ C) {          // [16,V]
  __shared__ _Float16 aLDS[CB * CH];                           // 8 KB
  int tid = threadIdx.x;                                       // 256 thr = 8 waves

  // Async-stage A: 8192 bytes = 512 x b128, 2 per thread.
  {
    const char* g = (const char*)A;
#pragma unroll
    for (int r = 0; r < 2; ++r) {
      int i = tid + r * 256;                 // b128 chunk index 0..511
      unsigned lds = (unsigned)(uintptr_t)(&aLDS[0]) + (unsigned)(i * 16);
      const char* ga = g + i * 16;
      asm volatile("global_load_async_to_lds_b128 %0, %1, off"
                   :: "v"(lds), "v"(ga) : "memory");
    }
    asm volatile("s_wait_asynccnt 0" ::: "memory");
  }
  __syncthreads();

  int wave = tid >> 5, lane = tid & 31;
  int n0 = (blockIdx.x * 8 + wave) * 16;
  if (n0 >= CV) return;

  float bv = bias[n0 + (lane & 15)];
  v8f acc;
#pragma unroll
  for (int j = 0; j < 8; ++j) acc[j] = bv;

  const _Float16* bp = Bw + (size_t)n0 * CH;
  __builtin_prefetch(bp + (size_t)16 * CH, 0, 1);   // next tile's B rows (L2)

  for (int k = 0; k < CH; k += 32) {
    v16h a = frag_load(aLDS + k, CH, lane);
    v16h b = frag_load(bp + k, CH, lane);
    acc = wmma_acc(a, b, acc);
  }

  int n = lane & 15;
  int mh = (lane & 16) ? 8 : 0;
#pragma unroll
  for (int j = 0; j < 8; ++j) {
    int m = mh + j;
    C[(size_t)m * CV + n0 + n] = acc[j];
  }
}

// ---------------------------------------------------------------------------
// Utility conversions / gathers
// ---------------------------------------------------------------------------
__global__ void k_f32_to_f16(const float* __restrict__ s, _Float16* __restrict__ d,
                             int n) {
  int i = blockIdx.x * blockDim.x + threadIdx.x;
  if (i < n) d[i] = (_Float16)s[i];
}

__global__ void k_embed_f16(const int* __restrict__ ids,
                            const float* __restrict__ emb,
                            _Float16* __restrict__ dst, int ntok) {
  int i = blockIdx.x * blockDim.x + threadIdx.x;
  if (i >= ntok * CE) return;
  int tkn = i / CE, c = i % CE;
  dst[i] = (_Float16)emb[(size_t)ids[tkn] * CE + c];
}

__global__ void k_init(const int* __restrict__ article, int* __restrict__ dec_in,
                       float* __restrict__ cov) {
  int i = blockIdx.x * blockDim.x + threadIdx.x;
  if (i < CB) dec_in[i] = article[(size_t)i * CS];
  if (i < CB * CS) cov[i] = 0.f;
}

// ---------------------------------------------------------------------------
// Encoder GRU scan. One block per direction (grid.x = 2), 512 threads =
// 16 waves; wave i owns gate-triple columns [16i,16i+16) of the 256-wide h.
// Hidden state ping-pongs in LDS as f16; gi = x@Wih.T + bih precomputed (f16).
// ---------------------------------------------------------------------------
__global__ void k_gru_scan(const _Float16* __restrict__ giF,
                           const _Float16* __restrict__ giB,
                           const _Float16* __restrict__ WhhF,
                           const _Float16* __restrict__ WhhB,
                           const float* __restrict__ bhhF,
                           const float* __restrict__ bhhB,
                           float* __restrict__ enc_out,      // [B*S, 2H]
                           _Float16* __restrict__ hcat16) {  // [B, 2H]
  int dir = blockIdx.x;
  const _Float16* gi  = dir ? giB : giF;
  const _Float16* Whh = dir ? WhhB : WhhF;
  const float*    bhh = dir ? bhhB : bhhF;

  __shared__ _Float16 hbuf[2][CB * CH];
  int tid = threadIdx.x;
  for (int i = tid; i < CB * CH; i += blockDim.x) hbuf[0][i] = (_Float16)0.f;
  __syncthreads();

  int wave = tid >> 5, lane = tid & 31;
  int n0 = wave * 16;
  int nn = lane & 15;
  int mh = (lane & 16) ? 8 : 0;
  float br = bhh[0 * CH + n0 + nn];
  float bz = bhh[1 * CH + n0 + nn];
  float bn = bhh[2 * CH + n0 + nn];

  int cur = 0;
  for (int step = 0; step < CS; ++step) {
    int tau = dir ? (CS - 1 - step) : step;
    const _Float16* hin = &hbuf[cur][0];

    v8f accR, accZ, accN;
#pragma unroll
    for (int j = 0; j < 8; ++j) { accR[j] = br; accZ[j] = bz; accN[j] = bn; }

    for (int k = 0; k < CH; k += 32) {
      v16h a = frag_load(hin + k, CH, lane);
      accR = wmma_acc(a, frag_load(Whh + (size_t)(0 * CH + n0) * CH + k, CH, lane), accR);
      accZ = wmma_acc(a, frag_load(Whh + (size_t)(1 * CH + n0) * CH + k, CH, lane), accZ);
      accN = wmma_acc(a, frag_load(Whh + (size_t)(2 * CH + n0) * CH + k, CH, lane), accN);
    }

#pragma unroll
    for (int j = 0; j < 8; ++j) {
      int m = mh + j;               // batch row
      int c = n0 + nn;              // hidden column
      size_t girow = ((size_t)m * CS + tau) * G3;
      float gir = (float)gi[girow + 0 * CH + c];
      float giz = (float)gi[girow + 1 * CH + c];
      float gin = (float)gi[girow + 2 * CH + c];
      float r = sigmf(gir + accR[j]);
      float z = sigmf(giz + accZ[j]);
      float nv = tanhf(gin + r * accN[j]);
      float hold = (float)hin[m * CH + c];
      float hnew = (1.f - z) * nv + z * hold;
      hbuf[cur ^ 1][m * CH + c] = (_Float16)hnew;
      enc_out[((size_t)m * CS + tau) * (2 * CH) + dir * CH + c] = hnew;
      if (step == CS - 1) hcat16[m * (2 * CH) + dir * CH + c] = (_Float16)hnew;
    }
    __syncthreads();
    cur ^= 1;
  }
}

// ---------------------------------------------------------------------------
// Decoder GRU step (one block, 512 threads). Same wave-per-gate-triple trick.
// ---------------------------------------------------------------------------
__global__ void k_dec_step(const int* __restrict__ dec_in,
                           const float* __restrict__ emb,
                           const _Float16* __restrict__ Wih16,   // [768,128]
                           const _Float16* __restrict__ Whh16,   // [768,256]
                           const float* __restrict__ bih,
                           const float* __restrict__ bhh,
                           const _Float16* __restrict__ h_in16,  // [16,256]
                           _Float16* __restrict__ h_out16,
                           float* __restrict__ hidden_f32) {
  __shared__ _Float16 e16[CB * CE];
  int tid = threadIdx.x;
  for (int i = tid; i < CB * CE; i += blockDim.x) {
    int b = i >> 7, c = i & 127;
    e16[i] = (_Float16)emb[(size_t)dec_in[b] * CE + c];
  }
  __syncthreads();

  int wave = tid >> 5, lane = tid & 31;
  int n0 = wave * 16;
  int nn = lane & 15;
  int mh = (lane & 16) ? 8 : 0;

  v8f accR, accZ, accGin, accGhn;
  {
    float br = bih[0 * CH + n0 + nn] + bhh[0 * CH + n0 + nn];
    float bz = bih[1 * CH + n0 + nn] + bhh[1 * CH + n0 + nn];
    float bi = bih[2 * CH + n0 + nn];
    float bh = bhh[2 * CH + n0 + nn];
#pragma unroll
    for (int j = 0; j < 8; ++j) { accR[j] = br; accZ[j] = bz; accGin[j] = bi; accGhn[j] = bh; }
  }

  // input projections  (K = 128)
  for (int k = 0; k < CE; k += 32) {
    v16h a = frag_load(e16 + k, CE, lane);
    accR   = wmma_acc(a, frag_load(Wih16 + (size_t)(0 * CH + n0) * CE + k, CE, lane), accR);
    accZ   = wmma_acc(a, frag_load(Wih16 + (size_t)(1 * CH + n0) * CE + k, CE, lane), accZ);
    accGin = wmma_acc(a, frag_load(Wih16 + (size_t)(2 * CH + n0) * CE + k, CE, lane), accGin);
  }
  // recurrent projections (K = 256)
  for (int k = 0; k < CH; k += 32) {
    v16h a = frag_load(h_in16 + k, CH, lane);
    accR   = wmma_acc(a, frag_load(Whh16 + (size_t)(0 * CH + n0) * CH + k, CH, lane), accR);
    accZ   = wmma_acc(a, frag_load(Whh16 + (size_t)(1 * CH + n0) * CH + k, CH, lane), accZ);
    accGhn = wmma_acc(a, frag_load(Whh16 + (size_t)(2 * CH + n0) * CH + k, CH, lane), accGhn);
  }

#pragma unroll
  for (int j = 0; j < 8; ++j) {
    int m = mh + j;
    int c = n0 + nn;
    float r = sigmf(accR[j]);
    float z = sigmf(accZ[j]);
    float nv = tanhf(accGin[j] + r * accGhn[j]);
    float hold = (float)h_in16[m * CH + c];
    float hnew = (1.f - z) * nv + z * hold;
    h_out16[m * CH + c] = (_Float16)hnew;
    hidden_f32[m * CH + c] = hnew;
  }
}

// ---------------------------------------------------------------------------
// Attention score: scores[b,s] = v . tanh(enc_feats[b,s,:] + sbias[b,:]), mask.
// ---------------------------------------------------------------------------
__global__ void k_attn_scores(const float* __restrict__ enc_feats,
                              const float* __restrict__ sbias,
                              const float* __restrict__ attn_v,
                              const int* __restrict__ article,
                              float* __restrict__ scores) {
  int s = blockIdx.x * blockDim.x + threadIdx.x;
  int b = blockIdx.y;
  if (s >= CS) return;
  const float* f  = enc_feats + ((size_t)b * CS + s) * CH;
  const float* sb = sbias + (size_t)b * CH;
  float acc = 0.f;
  for (int h = 0; h < CH; ++h) acc += attn_v[h] * tanhf(f[h] + sb[h]);
  scores[b * CS + s] = (article[(size_t)b * CS + s] != 0) ? acc : -INFINITY;
}

// Masked softmax over S per row, coverage update, strided output writes.
__global__ void k_attn_softmax(const float* __restrict__ scores,
                               float* __restrict__ attn, float* __restrict__ cov,
                               float* __restrict__ attnOut,
                               float* __restrict__ covOut, int t) {
  int b = threadIdx.x >> 5, lane = threadIdx.x & 31;
  float mx = -INFINITY;
  for (int s = lane; s < CS; s += 32) mx = fmaxf(mx, scores[b * CS + s]);
  for (int o = 16; o; o >>= 1) mx = fmaxf(mx, __shfl_xor(mx, o, 32));
  float sum = 0.f;
  for (int s = lane; s < CS; s += 32) sum += __expf(scores[b * CS + s] - mx);
  for (int o = 16; o; o >>= 1) sum += __shfl_xor(sum, o, 32);
  float inv = 1.f / sum;
  for (int s = lane; s < CS; s += 32) {
    float a = __expf(scores[b * CS + s] - mx) * inv;
    attn[b * CS + s] = a;
    float c = cov[b * CS + s] + a;
    cov[b * CS + s] = c;
    attnOut[((size_t)b * CS + s) * CT + t] = a;
    covOut[((size_t)b * CS + s) * CT + t] = c;
  }
}

// Context, p_gen, and hc=[hidden,context] assembly (one block, 512 threads).
__global__ void k_context(const float* __restrict__ attn,
                          const float* __restrict__ enc_out,
                          const float* __restrict__ hidden,
                          const float* __restrict__ emb,
                          const int* __restrict__ dec_in,
                          const float* __restrict__ whW, const float* __restrict__ whb,
                          const float* __restrict__ wsW, const float* __restrict__ wsb,
                          const float* __restrict__ wxW, const float* __restrict__ wxb,
                          float* __restrict__ context, float* __restrict__ pgen,
                          _Float16* __restrict__ hc16) {
  int tid = threadIdx.x;
  for (int i = tid; i < CB * 2 * CH; i += blockDim.x) {
    int b = i >> 9, k = i & 511;
    const float* a = attn + (size_t)b * CS;
    float acc = 0.f;
    for (int s = 0; s < CS; ++s) acc += a[s] * enc_out[((size_t)b * CS + s) * (2 * CH) + k];
    context[i] = acc;
    hc16[(size_t)b * G3 + CH + k] = (_Float16)acc;
  }
  for (int i = tid; i < CB * CH; i += blockDim.x) {
    int b = i >> 8, k = i & 255;
    hc16[(size_t)b * G3 + k] = (_Float16)hidden[i];
  }
  __syncthreads();
  if (tid < CB) {
    int b = tid;
    float acc = whb[0] + wsb[0] + wxb[0];
    for (int k = 0; k < 2 * CH; ++k) acc += whW[k] * context[b * 2 * CH + k];
    for (int k = 0; k < CH; ++k)     acc += wsW[k] * hidden[b * CH + k];
    const float* e = emb + (size_t)dec_in[b] * CE;
    for (int k = 0; k < CE; ++k)     acc += wxW[k] * e[k];
    pgen[b] = sigmf(acc);
  }
}

// Per-row max + exp-sum over the 50000 logits (one block per batch row).
__global__ void k_vocab_stats(const float* __restrict__ logits,
                              float* __restrict__ rowmax, float* __restrict__ rowsum) {
  __shared__ float red[512];
  int b = blockIdx.x, tid = threadIdx.x;
  const float* row = logits + (size_t)b * CV;
  float mx = -INFINITY;
  for (int v = tid; v < CV; v += 512) mx = fmaxf(mx, row[v]);
  red[tid] = mx; __syncthreads();
  for (int o = 256; o; o >>= 1) { if (tid < o) red[tid] = fmaxf(red[tid], red[tid + o]); __syncthreads(); }
  float m = red[0]; __syncthreads();
  float s = 0.f;
  for (int v = tid; v < CV; v += 512) s += __expf(row[v] - m);
  red[tid] = s; __syncthreads();
  for (int o = 256; o; o >>= 1) { if (tid < o) red[tid] += red[tid + o]; __syncthreads(); }
  if (tid == 0) { rowmax[b] = m; rowsum[b] = red[0]; }
}

// ext = concat(p_gen * softmax(logits), zeros(50)) written strided to d_out.
__global__ void k_final_dist(const float* __restrict__ logits,
                             const float* __restrict__ rowmax,
                             const float* __restrict__ rowsum,
                             const float* __restrict__ pgen,
                             float* __restrict__ out, int t) {
  int idx = blockIdx.x * blockDim.x + threadIdx.x;
  if (idx >= CB * CVE) return;
  int b = idx / CVE, v = idx % CVE;
  float val = 0.f;
  if (v < CV) val = pgen[b] * __expf(logits[(size_t)b * CV + v] - rowmax[b]) / rowsum[b];
  out[((size_t)b * CVE + v) * CT + t] = val;
}

// Scatter-add of the copy distribution (1-p_gen)*attn at text_ids_ext.
__global__ void k_scatter(const int* __restrict__ ids_ext,
                          const float* __restrict__ attn,
                          const float* __restrict__ pgen,
                          float* __restrict__ out, int t) {
  int i = blockIdx.x * blockDim.x + threadIdx.x;
  if (i >= CB * CS) return;
  int b = i / CS, s = i % CS;
  int tok = ids_ext[i];
  float w = (1.f - pgen[b]) * attn[b * CS + s];
  atomicAdd(&out[((size_t)b * CVE + tok) * CT + t], w);
}

// Two-phase argmax over the extended dist (first-max-index tie breaking).
__global__ void k_argmax_part(const float* __restrict__ out, int t,
                              float* __restrict__ pval, int* __restrict__ pidx) {
  __shared__ float sv[256];
  __shared__ int   si[256];
  int b = blockIdx.y, tid = threadIdx.x;
  int chunk = (CVE + gridDim.x - 1) / gridDim.x;
  int start = blockIdx.x * chunk;
  int end = start + chunk; if (end > CVE) end = CVE;
  float best = -INFINITY; int bix = CVE;
  for (int v = start + tid; v < end; v += 256) {
    float f = out[((size_t)b * CVE + v) * CT + t];
    if (f > best) { best = f; bix = v; }
  }
  sv[tid] = best; si[tid] = bix; __syncthreads();
  for (int o = 128; o; o >>= 1) {
    if (tid < o) {
      if (sv[tid + o] > sv[tid] || (sv[tid + o] == sv[tid] && si[tid + o] < si[tid])) {
        sv[tid] = sv[tid + o]; si[tid] = si[tid + o];
      }
    }
    __syncthreads();
  }
  if (tid == 0) { pval[b * gridDim.x + blockIdx.x] = sv[0]; pidx[b * gridDim.x + blockIdx.x] = si[0]; }
}

__global__ void k_argmax_fin(const float* __restrict__ pval,
                             const int* __restrict__ pidx,
                             int* __restrict__ dec_in) {
  __shared__ float sv[CB][64];
  __shared__ int   si[CB][64];
  int x = threadIdx.x, b = threadIdx.y;
  sv[b][x] = pval[b * 64 + x]; si[b][x] = pidx[b * 64 + x];
  __syncthreads();
  for (int o = 32; o; o >>= 1) {
    if (x < o) {
      if (sv[b][x + o] > sv[b][x] || (sv[b][x + o] == sv[b][x] && si[b][x + o] < si[b][x])) {
        sv[b][x] = sv[b][x + o]; si[b][x] = si[b][x + o];
      }
    }
    __syncthreads();
  }
  if (x == 0) { int top = si[b][0]; dec_in[b] = (top >= CV) ? 0 : top; }
}

// ---------------------------------------------------------------------------
// Host side
// ---------------------------------------------------------------------------
static inline void gemm(hipStream_t st, const _Float16* A, int lda,
                        const _Float16* Bw, int ldb, const float* bias,
                        float* C, int ldc, _Float16* D16, int ld16,
                        int M, int N, int K, int act) {
  dim3 g((N / 16 + 3) / 4, M / 16);
  k_gemm<<<g, 128, 0, st>>>(A, lda, Bw, ldb, bias, C, ldc, D16, ld16, M, N, K, act);
}

static inline void conv16(hipStream_t st, const float* s, _Float16* d, int n) {
  k_f32_to_f16<<<(n + 255) / 256, 256, 0, st>>>(s, d, n);
}

extern "C" void kernel_launch(void* const* d_in, const int* in_sizes, int n_in,
                              void* d_out, int out_size, void* d_ws, size_t ws_size,
                              hipStream_t stream) {
  (void)in_sizes; (void)n_in; (void)out_size; (void)ws_size;
  // ----- inputs (setup_inputs order; params dict flattened in insertion order)
  const int*   article  = (const int*)d_in[0];
  const int*   ids_ext  = (const int*)d_in[1];
  // d_in[2] summary_tensor, d_in[3] max_oov : unused by the forward pass
  const float* emb      = (const float*)d_in[4];
  const float* encWihF  = (const float*)d_in[5];
  const float* encWhhF  = (const float*)d_in[6];
  const float* encBihF  = (const float*)d_in[7];
  const float* encBhhF  = (const float*)d_in[8];
  const float* encWihB  = (const float*)d_in[9];
  const float* encWhhB  = (const float*)d_in[10];
  const float* encBihB  = (const float*)d_in[11];
  const float* encBhhB  = (const float*)d_in[12];
  const float* linW     = (const float*)d_in[13];
  const float* linB     = (const float*)d_in[14];
  const float* decWih   = (const float*)d_in[15];
  const float* decWhh   = (const float*)d_in[16];
  const float* decBih   = (const float*)d_in[17];
  const float* decBhh   = (const float*)d_in[18];
  const float* attnWh   = (const float*)d_in[19];
  const float* attnBh   = (const float*)d_in[20];
  const float* attnWs   = (const float*)d_in[21];
  const float* attnBs   = (const float*)d_in[22];
  const float* attnV    = (const float*)d_in[23];
  const float* vW       = (const float*)d_in[24];
  const float* vB       = (const float*)d_in[25];
  const float* vhatW    = (const float*)d_in[26];
  const float* vhatB    = (const float*)d_in[27];
  const float* whW      = (const float*)d_in[28];
  const float* whB      = (const float*)d_in[29];
  const float* wsW      = (const float*)d_in[30];
  const float* wsB      = (const float*)d_in[31];
  const float* wxW      = (const float*)d_in[32];
  const float* wxB      = (const float*)d_in[33];

  float* out      = (float*)d_out;
  float* attnOut  = out + (size_t)CB * CVE * CT;
  float* covOut   = attnOut + (size_t)CB * CS * CT;

  // ----- workspace arena
  char* base = (char*)d_ws;
  size_t off = 0;
  auto alloc = [&](size_t bytes) -> void* {
    void* p = base + off;
    off = (off + bytes + 255) & ~(size_t)255;
    return p;
  };
  _Float16* xf16      = (_Float16*)alloc((size_t)CB * CS * CE * 2);
  _Float16* wihF16    = (_Float16*)alloc((size_t)G3 * CE * 2);
  _Float16* whhF16    = (_Float16*)alloc((size_t)G3 * CH * 2);
  _Float16* wihB16    = (_Float16*)alloc((size_t)G3 * CE * 2);
  _Float16* whhB16    = (_Float16*)alloc((size_t)G3 * CH * 2);
  _Float16* linW16    = (_Float16*)alloc((size_t)CH * 2 * CH * 2);
  _Float16* decWih16  = (_Float16*)alloc((size_t)G3 * CE * 2);
  _Float16* decWhh16  = (_Float16*)alloc((size_t)G3 * CH * 2);
  _Float16* attnWh16  = (_Float16*)alloc((size_t)CH * 2 * CH * 2);
  _Float16* attnWs16  = (_Float16*)alloc((size_t)CH * CH * 2);
  _Float16* vW16      = (_Float16*)alloc((size_t)CH * G3 * 2);
  _Float16* vhat16    = (_Float16*)alloc((size_t)CV * CH * 2);
  _Float16* giF16     = (_Float16*)alloc((size_t)CB * CS * G3 * 2);
  _Float16* giB16     = (_Float16*)alloc((size_t)CB * CS * G3 * 2);
  float*    enc_out   = (float*)alloc((size_t)CB * CS * 2 * CH * 4);
  _Float16* encout16  = (_Float16*)alloc((size_t)CB * CS * 2 * CH * 2);
  _Float16* hcat16    = (_Float16*)alloc((size_t)CB * 2 * CH * 2);
  float*    hidden0f  = (float*)alloc((size_t)CB * CH * 4);
  float*    encfeats  = (float*)alloc((size_t)CB * CS * CH * 4);
  _Float16* h16       = (_Float16*)alloc((size_t)2 * CB * CH * 2);   // ping-pong
  float*    hiddenf   = (float*)alloc((size_t)CB * CH * 4);
  float*    sbias     = (float*)alloc((size_t)CB * CH * 4);
  float*    scores    = (float*)alloc((size_t)CB * CS * 4);
  float*    attnbuf   = (float*)alloc((size_t)CB * CS * 4);
  float*    covbuf    = (float*)alloc((size_t)CB * CS * 4);
  float*    context   = (float*)alloc((size_t)CB * 2 * CH * 4);
  _Float16* hc16      = (_Float16*)alloc((size_t)CB * G3 * 2);
  _Float16* vtmp16    = (_Float16*)alloc((size_t)CB * CH * 2);
  float*    logits    = (float*)alloc((size_t)CB * CV * 4);
  float*    rowmax    = (float*)alloc(CB * 4);
  float*    rowsum    = (float*)alloc(CB * 4);
  float*    pgen      = (float*)alloc(CB * 4);
  int*      dec_in    = (int*)alloc(CB * 4);
  float*    pval      = (float*)alloc((size_t)CB * 64 * 4);
  int*      pidx      = (int*)alloc((size_t)CB * 64 * 4);

  // ----- one-time f32 -> f16 weight conversion (L2-resident afterwards)
  conv16(stream, encWihF, wihF16, G3 * CE);
  conv16(stream, encWhhF, whhF16, G3 * CH);
  conv16(stream, encWihB, wihB16, G3 * CE);
  conv16(stream, encWhhB, whhB16, G3 * CH);
  conv16(stream, linW,    linW16, CH * 2 * CH);
  conv16(stream, decWih,  decWih16, G3 * CE);
  conv16(stream, decWhh,  decWhh16, G3 * CH);
  conv16(stream, attnWh,  attnWh16, CH * 2 * CH);
  conv16(stream, attnWs,  attnWs16, CH * CH);
  conv16(stream, vW,      vW16, CH * G3);
  conv16(stream, vhatW,   vhat16, CV * CH);

  // ----- encoder
  k_embed_f16<<<(CB * CS * CE + 255) / 256, 256, 0, stream>>>(article, emb, xf16, CB * CS);
  // gi = x @ Wih.T + bih for both directions (stored f16)
  gemm(stream, xf16, CE, wihF16, CE, encBihF, nullptr, 0, giF16, G3, CB * CS, G3, CE, 0);
  gemm(stream, xf16, CE, wihB16, CE, encBihB, nullptr, 0, giB16, G3, CB * CS, G3, CE, 0);
  // both direction scans run concurrently (grid.x = 2)
  k_gru_scan<<<2, 512, 0, stream>>>(giF16, giB16, whhF16, whhB16, encBhhF, encBhhB,
                                    enc_out, hcat16);
  conv16(stream, enc_out, encout16, CB * CS * 2 * CH);
  // hidden0 = leaky_relu([hF,hB] @ lin.T + b) -> decoder h slot 0 (f16)
  gemm(stream, hcat16, 2 * CH, linW16, 2 * CH, linB, hidden0f, CH, h16, CH,
       CB, CH, 2 * CH, 1);
  // enc_feats = enc_out @ attn_Wh.T + bh
  gemm(stream, encout16, 2 * CH, attnWh16, 2 * CH, attnBh, encfeats, CH, nullptr, 0,
       CB * CS, CH, 2 * CH, 0);

  k_init<<<(CB * CS + 255) / 256, 256, 0, stream>>>(article, dec_in, covbuf);

  // ----- decoder scan (T = 64 sequential steps)
  for (int t = 0; t < CT; ++t) {
    _Float16* h_in  = h16 + (size_t)(t & 1) * CB * CH;
    _Float16* h_out = h16 + (size_t)((t + 1) & 1) * CB * CH;

    k_dec_step<<<1, 512, 0, stream>>>(dec_in, emb, decWih16, decWhh16, decBih,
                                      decBhh, h_in, h_out, hiddenf);
    // sbias = hidden @ attn_Ws.T + bs
    gemm(stream, h_out, CH, attnWs16, CH, attnBs, sbias, CH, nullptr, 0,
         CB, CH, CH, 0);
    k_attn_scores<<<dim3((CS + 127) / 128, CB), 128, 0, stream>>>(
        encfeats, sbias, attnV, article, scores);
    k_attn_softmax<<<1, 512, 0, stream>>>(scores, attnbuf, covbuf, attnOut, covOut, t);
    k_context<<<1, 512, 0, stream>>>(attnbuf, enc_out, hiddenf, emb, dec_in,
                                     whW, whB, wsW, wsB, wxW, wxB,
                                     context, pgen, hc16);
    // vtmp = hc @ v_W.T + v_b  (f16 out for the vocab GEMM)
    gemm(stream, hc16, G3, vW16, G3, vB, nullptr, 0, vtmp16, CH, CB, CH, G3, 0);
    // logits = vtmp @ vhat_W.T + vhat_b  -- the hot 26-GFLOP GEMM, L2 resident,
    // A staged via CDNA5 async-DMA to LDS.
    k_gemm_vocab<<<(CV / 16 + 7) / 8, 256, 0, stream>>>(vtmp16, vhat16, vhatB, logits);
    k_vocab_stats<<<CB, 512, 0, stream>>>(logits, rowmax, rowsum);
    k_final_dist<<<(CB * CVE + 255) / 256, 256, 0, stream>>>(
        logits, rowmax, rowsum, pgen, out, t);
    k_scatter<<<(CB * CS + 255) / 256, 256, 0, stream>>>(ids_ext, attnbuf, pgen, out, t);
    k_argmax_part<<<dim3(64, CB), 256, 0, stream>>>(out, t, pval, pidx);
    k_argmax_fin<<<1, dim3(64, CB), 0, stream>>>(pval, pidx, dec_in);
  }
}
